// LongformerForMultiLabelRegression_73830487818950
// MI455X (gfx1250) — matrix operand order repaired and checked by reading, not compiled, not run on hardware
//
#include <hip/hip_runtime.h>
#include <hip/hip_bf16.h>
#include <math.h>

// ---------------------------------------------------------------------------
// Longformer forward for MI455X (gfx1250, wave32, WMMA).
// - All matmuls: v_wmma_f32_16x16x32_f16, fp32 accumulate.
// - GEMM: double-buffered LDS pipeline (reg-prefetch next tile during WMMA).
// - Attention context: f16 probs copied global->LDS via
//   GLOBAL_LOAD_ASYNC_TO_LDS_B128 (ASYNCcnt) -- no VGPR round trip.
// ---------------------------------------------------------------------------

#define BB   2
#define SS   1024
#define DD   768
#define HH   12
#define DHH  64
#define FFD  3072
#define LL   12
#define BLKK 256
#define NBB  4
#define SCW  769          // 1 global col + 768 local cols
#define NEGV (-1000000000.0f)
#define EPSV 1e-5f

typedef __attribute__((ext_vector_type(16))) _Float16 v16h;
typedef __attribute__((ext_vector_type(8)))  float    v8f;

// ------------------------- WMMA fragment helpers ---------------------------
// A (16x32 f16): lanes 0-15 hold M=0-15 / K=0-7,16-23 ; lanes 16-31 K=8-15,24-31
__device__ inline v16h frag_a(const _Float16* tile, int row0, int stride, int lane) {
  int m  = row0 + (lane & 15);
  int kb = (lane >> 4) * 8;
  const _Float16* p = tile + m * stride;
  v16h a;
#pragma unroll
  for (int e = 0; e < 8; ++e) {
    a[e]     = p[kb + e];
    a[e + 8] = p[16 + kb + e];
  }
  return a;
}

// B (32x16 f16, tile stored [K][N]): lanes 0-15 hold N=0-15 / K=0-7,16-23 ; lanes 16-31 K=8-15,24-31
__device__ inline v16h frag_b(const _Float16* tile, int col0, int stride, int lane) {
  const _Float16* p = tile + col0 + (lane & 15);
  int kb = (lane >> 4) * 8;
  v16h b;
#pragma unroll
  for (int e = 0; e < 8; ++e) {
    b[e]     = p[(kb + e) * stride];
    b[e + 8] = p[(16 + kb + e) * stride];
  }
  return b;
}

__device__ inline v8f zero8() {
  v8f z = {0.f, 0.f, 0.f, 0.f, 0.f, 0.f, 0.f, 0.f};
  return z;
}

// ------------------------- block reductions --------------------------------
__device__ inline float breduce_sum(float v, float* red) {
  int tid = threadIdx.x;
  red[tid] = v; __syncthreads();
#pragma unroll
  for (int o = 128; o > 0; o >>= 1) {
    if (tid < o) red[tid] += red[tid + o];
    __syncthreads();
  }
  float r = red[0]; __syncthreads();
  return r;
}

__device__ inline float breduce_max(float v, float* red) {
  int tid = threadIdx.x;
  red[tid] = v; __syncthreads();
#pragma unroll
  for (int o = 128; o > 0; o >>= 1) {
    if (tid < o) red[tid] = fmaxf(red[tid], red[tid + o]);
    __syncthreads();
  }
  float r = red[0]; __syncthreads();
  return r;
}

// ===========================================================================
// Generic GEMM: C[M,N] = act((A[M,K] @ B[K,N] + bias) * alpha)
// Macro tile 128x128, 8 waves of 32x64 (2x4 WMMA accs), K step 32.
// Double-buffered LDS: next tile prefetched into registers during WMMA.
// ===========================================================================
__global__ void __launch_bounds__(256) gemm_kernel(
    const float* __restrict__ A, const float* __restrict__ B,
    const float* __restrict__ bias, float* __restrict__ C,
    int M, int N, int K, float alpha, int fuse /*0=none,1=gelu*/) {
  __shared__ _Float16 As[2][128 * 36];
  __shared__ _Float16 Bs[2][32 * 132];
  int tid = threadIdx.x;
  int lane = tid & 31;
  int wid = tid >> 5;
  int wm = wid & 3;      // 4 waves over M (32 rows each)
  int wn = wid >> 2;     // 2 waves over N (64 cols each)
  int m0 = blockIdx.y * 128;
  int n0 = blockIdx.x * 128;

  int ar = tid >> 1, ac = (tid & 1) * 16;   // A: 128 rows x 32 cols
  int br = tid >> 3, bc = (tid & 7) * 16;   // B: 32 rows x 128 cols
  const float* aptr = A + (size_t)(m0 + ar) * K + ac;
  const float* bptr = B + (size_t)br * N + n0 + bc;

  float4 ra[4], rb[4];
  auto ldA = [&](int k0) {
    const float4* s = (const float4*)(aptr + k0);
#pragma unroll
    for (int e = 0; e < 4; ++e) ra[e] = s[e];
  };
  auto ldB = [&](int k0) {
    const float4* s = (const float4*)(bptr + (size_t)k0 * N);
#pragma unroll
    for (int e = 0; e < 4; ++e) rb[e] = s[e];
  };
  auto stA = [&](int buf) {
    _Float16* d = As[buf] + ar * 36 + ac;
#pragma unroll
    for (int e = 0; e < 4; ++e) {
      d[4 * e + 0] = (_Float16)ra[e].x;
      d[4 * e + 1] = (_Float16)ra[e].y;
      d[4 * e + 2] = (_Float16)ra[e].z;
      d[4 * e + 3] = (_Float16)ra[e].w;
    }
  };
  auto stB = [&](int buf) {
    _Float16* d = Bs[buf] + br * 132 + bc;
#pragma unroll
    for (int e = 0; e < 4; ++e) {
      d[4 * e + 0] = (_Float16)rb[e].x;
      d[4 * e + 1] = (_Float16)rb[e].y;
      d[4 * e + 2] = (_Float16)rb[e].z;
      d[4 * e + 3] = (_Float16)rb[e].w;
    }
  };

  v8f acc[2][4];
#pragma unroll
  for (int i = 0; i < 2; ++i)
#pragma unroll
    for (int j = 0; j < 4; ++j) acc[i][j] = zero8();

  // prologue: fill buffer 0
  ldA(0); ldB(0); stA(0); stB(0);
  __syncthreads();

  int KT = K >> 5;
  for (int kt = 0; kt < KT; ++kt) {
    int cur = kt & 1;
    bool pre = (kt + 1) < KT;
    if (pre) { ldA((kt + 1) * 32); ldB((kt + 1) * 32); }   // overlap with WMMA

    v16h af[2], bf[4];
#pragma unroll
    for (int i = 0; i < 2; ++i) af[i] = frag_a(As[cur], wm * 32 + i * 16, 36, lane);
#pragma unroll
    for (int j = 0; j < 4; ++j) bf[j] = frag_b(Bs[cur], wn * 64 + j * 16, 132, lane);
#pragma unroll
    for (int i = 0; i < 2; ++i)
#pragma unroll
      for (int j = 0; j < 4; ++j)
        acc[i][j] = __builtin_amdgcn_wmma_f32_16x16x32_f16(
            false, af[i], false, bf[j], (short)0, acc[i][j], false, false);

    if (pre) { stA(cur ^ 1); stB(cur ^ 1); }
    __syncthreads();
  }

#pragma unroll
  for (int i = 0; i < 2; ++i)
#pragma unroll
    for (int j = 0; j < 4; ++j) {
      int col = n0 + wn * 64 + j * 16 + (lane & 15);
      int row = m0 + wm * 32 + i * 16 + (lane >> 4) * 8;
      float bc2 = bias ? bias[col] : 0.0f;
#pragma unroll
      for (int r = 0; r < 8; ++r) {
        float v = (acc[i][j][r] + bc2) * alpha;
        if (fuse == 1) v = 0.5f * v * (1.0f + erff(v * 0.70710678118654752f));
        C[(size_t)(row + r) * N + col] = v;
      }
    }
}

// ===========================================================================
// Embedding gather + LayerNorm  (one block per token row)
// ===========================================================================
__global__ void __launch_bounds__(256) embed_ln_kernel(
    const int* __restrict__ ids, const float* __restrict__ we,
    const float* __restrict__ pe, const float* __restrict__ te,
    const float* __restrict__ g, const float* __restrict__ be,
    float* __restrict__ x) {
  __shared__ float red[256];
  int row = blockIdx.x;
  int s = row % SS;
  int tid = threadIdx.x;
  int tok = ids[row];
  float vv[3];
#pragma unroll
  for (int rep = 0; rep < 3; ++rep) {
    int d = tid + rep * 256;
    vv[rep] = we[(size_t)tok * DD + d] + pe[(size_t)(s + 2) * DD + d] + te[d];
  }
  float mean = breduce_sum(vv[0] + vv[1] + vv[2], red) * (1.0f / DD);
  float sq = 0.f;
#pragma unroll
  for (int rep = 0; rep < 3; ++rep) { float dlt = vv[rep] - mean; sq += dlt * dlt; }
  float var = breduce_sum(sq, red) * (1.0f / DD);
  float isr = rsqrtf(var + EPSV);
#pragma unroll
  for (int rep = 0; rep < 3; ++rep) {
    int d = tid + rep * 256;
    x[(size_t)row * DD + d] = (vv[rep] - mean) * isr * g[d] + be[d];
  }
}

// ===========================================================================
// x = LayerNorm(x + a)   (one block per token row)
// ===========================================================================
__global__ void __launch_bounds__(256) add_ln_kernel(
    const float* __restrict__ xin, const float* __restrict__ a,
    const float* __restrict__ g, const float* __restrict__ be,
    float* __restrict__ xout) {
  __shared__ float red[256];
  int row = blockIdx.x;
  int tid = threadIdx.x;
  float vv[3];
#pragma unroll
  for (int rep = 0; rep < 3; ++rep) {
    int d = tid + rep * 256;
    vv[rep] = xin[(size_t)row * DD + d] + a[(size_t)row * DD + d];
  }
  float mean = breduce_sum(vv[0] + vv[1] + vv[2], red) * (1.0f / DD);
  float sq = 0.f;
#pragma unroll
  for (int rep = 0; rep < 3; ++rep) { float dlt = vv[rep] - mean; sq += dlt * dlt; }
  float var = breduce_sum(sq, red) * (1.0f / DD);
  float isr = rsqrtf(var + EPSV);
#pragma unroll
  for (int rep = 0; rep < 3; ++rep) {
    int d = tid + rep * 256;
    xout[(size_t)row * DD + d] = (vv[rep] - mean) * isr * g[d] + be[d];
  }
}

// ===========================================================================
// Global score column: scores[...,0] = q[b, s, h, :] . k[b, 0, h, :]
// ===========================================================================
__global__ void __launch_bounds__(256) attn_sg_kernel(
    const float* __restrict__ q, const float* __restrict__ k,
    float* __restrict__ scores) {
  int blk = blockIdx.x;
  int b = blk / (NBB * HH);
  int n = (blk / HH) % NBB;
  int h = blk % HH;
  int tid = threadIdx.x;
  const float* qp = q + (size_t)(b * SS + n * BLKK + tid) * DD + h * DHH;
  const float* kp = k + (size_t)(b * SS) * DD + h * DHH;
  float acc = 0.f;
#pragma unroll 8
  for (int d = 0; d < DHH; ++d) acc += qp[d] * kp[d];
  size_t srow = ((size_t)((b * HH + h) * NBB + n)) * BLKK + tid;
  scores[srow * SCW] = acc;
}

// ===========================================================================
// Banded scores via WMMA: per (b, n, h): [256 x 64] @ [64 x 768(keys)]
// Structural Longformer mask fused in the epilogue.
// ===========================================================================
__global__ void __launch_bounds__(256) attn_scores_kernel(
    const float* __restrict__ q, const float* __restrict__ k,
    const int* __restrict__ amask, float* __restrict__ scores) {
  int blk = blockIdx.x;
  int b = blk / (NBB * HH);
  int n = (blk / HH) % NBB;
  int h = blk % HH;
  int tid = threadIdx.x;
  int lane = tid & 31;
  int wid = tid >> 5;
  __shared__ _Float16 Qs[BLKK * 68];
  __shared__ _Float16 Ks[DHH * 68];

  // stage the full Q block (256 x 64) once
  {
    const float* src = q + (size_t)(b * SS + n * BLKK + tid) * DD + h * DHH;
    _Float16* dst = Qs + tid * 68;
#pragma unroll
    for (int e = 0; e < 16; ++e) {
      float4 f = ((const float4*)src)[e];
      dst[e * 4 + 0] = (_Float16)f.x;
      dst[e * 4 + 1] = (_Float16)f.y;
      dst[e * 4 + 2] = (_Float16)f.z;
      dst[e * 4 + 3] = (_Float16)f.w;
    }
  }
  size_t srow = ((size_t)((b * HH + h) * NBB + n)) * BLKK;

  for (int c = 0; c < 12; ++c) {           // 12 chunks of 64 keys
    int j0 = c * 64;
    __syncthreads();
    // stage K chunk transposed: Ks[d][jj] = k3[j0+jj][d]
    {
      int jj = tid >> 2;
      int dbase = (tid & 3) * 16;
      int kpos = (n - 1) * BLKK + j0 + jj;
      if (kpos >= 0 && kpos < SS) {
        const float* src = k + (size_t)(b * SS + kpos) * DD + h * DHH + dbase;
#pragma unroll
        for (int e = 0; e < 16; ++e) Ks[(dbase + e) * 68 + jj] = (_Float16)src[e];
      } else {
#pragma unroll
        for (int e = 0; e < 16; ++e) Ks[(dbase + e) * 68 + jj] = (_Float16)0.0f;
      }
    }
    __syncthreads();

    v8f acc[2][4];
#pragma unroll
    for (int i = 0; i < 2; ++i)
#pragma unroll
      for (int j = 0; j < 4; ++j) acc[i][j] = zero8();

#pragma unroll
    for (int kk = 0; kk < 2; ++kk) {       // K = 64 -> 2 WMMA K-steps
      v16h af[2], bf[4];
#pragma unroll
      for (int i = 0; i < 2; ++i) af[i] = frag_a(Qs + kk * 32, wid * 32 + i * 16, 68, lane);
#pragma unroll
      for (int j = 0; j < 4; ++j) bf[j] = frag_b(Ks + kk * 32 * 68, j * 16, 68, lane);
#pragma unroll
      for (int i = 0; i < 2; ++i)
#pragma unroll
        for (int j = 0; j < 4; ++j)
          acc[i][j] = __builtin_amdgcn_wmma_f32_16x16x32_f16(
              false, af[i], false, bf[j], (short)0, acc[i][j], false, false);
    }

    // masked store
#pragma unroll
    for (int i = 0; i < 2; ++i)
#pragma unroll
      for (int j = 0; j < 4; ++j) {
        int col = j * 16 + (lane & 15);
        int jg = j0 + col;
        int kpos = (n - 1) * BLKK + jg;
        int kc = kpos < 0 ? 0 : (kpos > SS - 1 ? SS - 1 : kpos);
        bool valid = (kpos >= 0) && (kpos < SS) && (kpos != 0);
        bool pad = amask[b * SS + kc] != 0;
#pragma unroll
        for (int r = 0; r < 8; ++r) {
          int qq = wid * 32 + i * 16 + (lane >> 4) * 8 + r;
          int delta = jg - BLKK - qq;
          bool win = (delta >= -BLKK) && (delta <= BLKK);
          float v = (win && valid && pad) ? acc[i][j][r] : NEGV;
          scores[(srow + qq) * SCW + 1 + jg] = v;
        }
      }
  }
}

// ===========================================================================
// Row softmax over 769 entries; emits p0 (fp32) + local probs (f16 for WMMA)
// ===========================================================================
__global__ void __launch_bounds__(256) softmax_kernel(
    const float* __restrict__ scores, float* __restrict__ p0,
    _Float16* __restrict__ ph) {
  __shared__ float red[256];
  int row = blockIdx.x;
  int tid = threadIdx.x;
  const float* sp = scores + (size_t)row * SCW;
  float e0 = sp[tid], e1 = sp[tid + 256], e2 = sp[tid + 512];
  float e3 = (tid == 0) ? sp[768] : NEGV;
  float mx = breduce_max(fmaxf(fmaxf(e0, e1), fmaxf(e2, e3)), red);
  float x0 = __expf(e0 - mx), x1 = __expf(e1 - mx), x2 = __expf(e2 - mx);
  float x3 = (tid == 0) ? __expf(e3 - mx) : 0.0f;
  float inv = 1.0f / breduce_sum(x0 + x1 + x2 + x3, red);
  _Float16* php = ph + (size_t)row * 768;
  if (tid == 0) p0[row] = x0 * inv;
  else          php[tid - 1] = (_Float16)(x0 * inv);
  php[tid + 255] = (_Float16)(x1 * inv);
  php[tid + 511] = (_Float16)(x2 * inv);
  if (tid == 0) php[767] = (_Float16)(x3 * inv);
}

// ===========================================================================
// Context via WMMA: per (b, n, h): [256 x 768(probs)] @ [768(keys) x 64] + p0*v0
// Probs are f16 in global: copied straight to LDS with ASYNC b128 loads.
// ===========================================================================
__global__ void __launch_bounds__(256) attn_out_kernel(
    const _Float16* __restrict__ ph, const float* __restrict__ v,
    const float* __restrict__ p0, float* __restrict__ ctx) {
  int blk = blockIdx.x;
  int b = blk / (NBB * HH);
  int n = (blk / HH) % NBB;
  int h = blk % HH;
  int tid = threadIdx.x;
  int lane = tid & 31;
  int wid = tid >> 5;
  __shared__ _Float16 As[BLKK * 40];       // stride 40 halves = 80 B (16B aligned)
  __shared__ _Float16 Bs[32 * 68];
  size_t prow = ((size_t)((b * HH + h) * NBB + n)) * BLKK;
  const _Float16* pbase = ph + prow * 768;

  v8f acc[2][4];
#pragma unroll
  for (int i = 0; i < 2; ++i)
#pragma unroll
    for (int j = 0; j < 4; ++j) acc[i][j] = zero8();

  for (int c = 0; c < 24; ++c) {           // K = 768 -> 24 WMMA K-steps
    int k0 = c * 32;
    __syncthreads();                       // prior compute done before rewrite
    // A tile: 256 rows x 32 halves, pure f16 copy -> async direct to LDS.
    // 4 transfers of 16B per thread (ASYNCcnt-tracked).
    {
      int sub = tid & 3;                   // 16B segment within the 64B row
#pragma unroll
      for (int rr = 0; rr < 4; ++rr) {
        int r = (tid >> 2) + rr * 64;
        const _Float16* gsrc = pbase + (size_t)r * 768 + k0 + sub * 8;
        unsigned laddr = (unsigned)(unsigned long long)(As + r * 40 + sub * 8);
        unsigned long long ga = (unsigned long long)gsrc;
        asm volatile("global_load_async_to_lds_b128 %0, %1, off"
                     :: "v"(laddr), "v"(ga)
                     : "memory");
      }
    }
    {   // B tile: Bs[kk][d] = v3[k0+kk][d]  (fp32 -> f16 via VGPRs)
      int kk = tid >> 3;
      int dbase = (tid & 7) * 8;
      int kpos = (n - 1) * BLKK + k0 + kk;
      if (kpos >= 0 && kpos < SS) {
        const float* src = v + (size_t)(b * SS + kpos) * DD + h * DHH + dbase;
#pragma unroll
        for (int e = 0; e < 8; ++e) Bs[kk * 68 + dbase + e] = (_Float16)src[e];
      } else {
#pragma unroll
        for (int e = 0; e < 8; ++e) Bs[kk * 68 + dbase + e] = (_Float16)0.0f;
      }
    }
    asm volatile("s_wait_asynccnt 0x0" ::: "memory");
    __syncthreads();
    v16h af[2], bf[4];
#pragma unroll
    for (int i = 0; i < 2; ++i) af[i] = frag_a(As, wid * 32 + i * 16, 40, lane);
#pragma unroll
    for (int j = 0; j < 4; ++j) bf[j] = frag_b(Bs, j * 16, 68, lane);
#pragma unroll
    for (int i = 0; i < 2; ++i)
#pragma unroll
      for (int j = 0; j < 4; ++j)
        acc[i][j] = __builtin_amdgcn_wmma_f32_16x16x32_f16(
            false, af[i], false, bf[j], (short)0, acc[i][j], false, false);
  }

#pragma unroll
  for (int i = 0; i < 2; ++i)
#pragma unroll
    for (int j = 0; j < 4; ++j) {
      int col = j * 16 + (lane & 15);
      float v0 = v[(size_t)(b * SS) * DD + h * DHH + col];
#pragma unroll
      for (int r = 0; r < 8; ++r) {
        int qq = wid * 32 + i * 16 + (lane >> 4) * 8 + r;
        float g = p0[prow + qq] * v0;
        ctx[(size_t)(b * SS + n * BLKK + qq) * DD + h * DHH + col] = acc[i][j][r] + g;
      }
    }
}

// ===========================================================================
// Global query projection (M=2, VALU): qg = (x[:,0] @ Wqg + bqg) * 0.125
// ===========================================================================
__global__ void __launch_bounds__(256) qg_kernel(
    const float* __restrict__ x, const float* __restrict__ W,
    const float* __restrict__ bias, float* __restrict__ qg) {
  int b = blockIdx.x;
  int tid = threadIdx.x;
  const float* xr = x + (size_t)(b * SS) * DD;
#pragma unroll
  for (int rep = 0; rep < 3; ++rep) {
    int d = tid + rep * 256;
    float acc = bias[d];
    for (int kk = 0; kk < DD; ++kk) acc += xr[kk] * W[(size_t)kk * DD + d];
    qg[b * DD + d] = acc * 0.125f;
  }
}

// ===========================================================================
// Global-token attention: softmax(qg . kg over S) @ vg -> ctx row 0
// One block per (b, h).
// ===========================================================================
__global__ void __launch_bounds__(256) global_attn_kernel(
    const float* __restrict__ qg, const float* __restrict__ kg,
    const float* __restrict__ vg, const int* __restrict__ amask,
    float* __restrict__ ctx) {
  __shared__ float sc[SS];
  __shared__ float red[256];
  __shared__ float qsh[DHH];
  __shared__ float part[4][DHH];
  int b = blockIdx.x / HH;
  int h = blockIdx.x % HH;
  int tid = threadIdx.x;
  if (tid < DHH) qsh[tid] = qg[b * DD + h * DHH + tid];
  __syncthreads();
  float lm = NEGV;
#pragma unroll
  for (int rep = 0; rep < 4; ++rep) {
    int s = tid + rep * 256;
    const float* kp = kg + (size_t)(b * SS + s) * DD + h * DHH;
    float a = 0.f;
#pragma unroll 8
    for (int d = 0; d < DHH; ++d) a += qsh[d] * kp[d];
    if (amask[b * SS + s] == 0) a += NEGV;
    sc[s] = a;
    lm = fmaxf(lm, a);
  }
  float mx = breduce_max(lm, red);
  float ls = 0.f;
#pragma unroll
  for (int rep = 0; rep < 4; ++rep) {
    int s = tid + rep * 256;
    float e = __expf(sc[s] - mx);
    sc[s] = e;
    ls += e;
  }
  float inv = 1.0f / breduce_sum(ls, red);   // barriers also publish sc[]
  int grp = tid >> 6;
  int dd = tid & 63;
  float accd = 0.f;
  for (int s = grp * 256; s < grp * 256 + 256; ++s)
    accd += sc[s] * vg[(size_t)(b * SS + s) * DD + h * DHH + dd];
  part[grp][dd] = accd;
  __syncthreads();
  if (tid < DHH)
    ctx[(size_t)(b * SS) * DD + h * DHH + tid] =
        (part[0][tid] + part[1][tid] + part[2][tid] + part[3][tid]) * inv;
}

// ===========================================================================
// Batch-norm over B=2 + MLP head (relu -> tanh). Single block.
// ===========================================================================
__global__ void __launch_bounds__(256) head_kernel(
    const float* __restrict__ x, const float* __restrict__ bn_g,
    const float* __restrict__ bn_b, const float* __restrict__ W1,
    const float* __restrict__ b1, const float* __restrict__ W2,
    const float* __restrict__ b2, float* __restrict__ out, int out_off) {
  __shared__ float hb[2][DD];
  __shared__ float red[256];
  int tid = threadIdx.x;
#pragma unroll
  for (int rep = 0; rep < 3; ++rep) {
    int d = tid + rep * 256;
    float a0 = x[(size_t)(0 * SS) * DD + d];
    float a1 = x[(size_t)(1 * SS) * DD + d];
    float mu = 0.5f * (a0 + a1);
    float d0 = a0 - mu, d1 = a1 - mu;
    float var = 0.5f * (d0 * d0 + d1 * d1);
    float sc = rsqrtf(var + EPSV) * bn_g[d];
    hb[0][d] = d0 * sc + bn_b[d];
    hb[1][d] = d1 * sc + bn_b[d];
  }
  __syncthreads();
  float hid0 = b1[tid], hid1 = b1[tid];
  for (int kk = 0; kk < DD; ++kk) {
    float w = W1[(size_t)kk * 256 + tid];
    hid0 += hb[0][kk] * w;
    hid1 += hb[1][kk] * w;
  }
  hid0 = fmaxf(hid0, 0.f);
  hid1 = fmaxf(hid1, 0.f);
  float w2 = W2[tid];
  float r0 = breduce_sum(hid0 * w2, red);
  float r1 = breduce_sum(hid1 * w2, red);
  if (tid == 0) {
    out[out_off + 0] = tanhf(r0 + b2[0]);
    out[out_off + 1] = tanhf(r1 + b2[0]);
  }
}

// ===========================================================================
// Host-side orchestration
// ===========================================================================
extern "C" void kernel_launch(void* const* d_in, const int* in_sizes, int n_in,
                              void* d_out, int out_size, void* d_ws, size_t ws_size,
                              hipStream_t stream) {
  (void)in_sizes; (void)n_in; (void)out_size; (void)ws_size;

  const int*   ids   = (const int*)d_in[0];
  const int*   amask = (const int*)d_in[1];
  // d_in[2] = global_attention_mask (structure is known: only token 0 global)
  const float* word  = (const float*)d_in[3];
  const float* pos   = (const float*)d_in[4];
  const float* type0 = (const float*)d_in[5];
  const float* eg    = (const float*)d_in[6];
  const float* eb    = (const float*)d_in[7];
  const float* Wq_a  = (const float*)d_in[8];
  const float* bq_a  = (const float*)d_in[9];
  const float* Wk_a  = (const float*)d_in[10];
  const float* bk_a  = (const float*)d_in[11];
  const float* Wv_a  = (const float*)d_in[12];
  const float* bv_a  = (const float*)d_in[13];
  const float* Wqg_a = (const float*)d_in[14];
  const float* bqg_a = (const float*)d_in[15];
  const float* Wkg_a = (const float*)d_in[16];
  const float* bkg_a = (const float*)d_in[17];
  const float* Wvg_a = (const float*)d_in[18];
  const float* bvg_a = (const float*)d_in[19];
  const float* Wo_a  = (const float*)d_in[20];
  const float* bo_a  = (const float*)d_in[21];
  const float* l1g_a = (const float*)d_in[22];
  const float* l1b_a = (const float*)d_in[23];
  const float* W1_a  = (const float*)d_in[24];
  const float* b1_a  = (const float*)d_in[25];
  const float* W2_a  = (const float*)d_in[26];
  const float* b2_a  = (const float*)d_in[27];
  const float* l2g_a = (const float*)d_in[28];
  const float* l2b_a = (const float*)d_in[29];
  const float* bn_g  = (const float*)d_in[30];
  const float* bn_b  = (const float*)d_in[31];
  const float* polW1 = (const float*)d_in[32];
  const float* polb1 = (const float*)d_in[33];
  const float* polW2 = (const float*)d_in[34];
  const float* polb2 = (const float*)d_in[35];
  const float* drgW1 = (const float*)d_in[36];
  const float* drgb1 = (const float*)d_in[37];
  const float* drgW2 = (const float*)d_in[38];
  const float* drgb2 = (const float*)d_in[39];

  // workspace carve-up (float elements)
  float* wsf = (float*)d_ws;
  size_t off = 0;
  const size_t NTOK = (size_t)BB * SS;               // 2048
  float* x    = wsf + off; off += NTOK * DD;
  float* qb   = wsf + off; off += NTOK * DD;
  float* kb   = wsf + off; off += NTOK * DD;
  float* vb   = wsf + off; off += NTOK * DD;
  float* kgb  = wsf + off; off += NTOK * DD;
  float* vgb  = wsf + off; off += NTOK * DD;
  float* ctx  = wsf + off; off += NTOK * DD;
  float* proj = wsf + off; off += NTOK * DD;
  float* ffh  = wsf + off; off += NTOK * FFD;
  float* ffy  = wsf + off; off += NTOK * DD;
  const size_t NROWS = (size_t)BB * HH * NBB * BLKK; // 24576 score rows
  float* scores = wsf + off; off += NROWS * SCW;
  float* p0     = wsf + off; off += NROWS;
  float* qg     = wsf + off; off += (size_t)BB * DD;
  _Float16* ph  = (_Float16*)(wsf + off); off += (NROWS * 768) / 2;

  const dim3 blk256(256);
  const dim3 gDxD(DD / 128, (int)(NTOK / 128));      // (6,16)
  const dim3 gDxFF(FFD / 128, (int)(NTOK / 128));    // (24,16)
  const int ABLKS = BB * NBB * HH;                   // 96

  // embeddings + LN
  embed_ln_kernel<<<(int)NTOK, blk256, 0, stream>>>(ids, word, pos, type0, eg, eb, x);

  for (int l = 0; l < LL; ++l) {
    const float* Wq  = Wq_a  + (size_t)l * DD * DD;  const float* bq  = bq_a  + (size_t)l * DD;
    const float* Wk  = Wk_a  + (size_t)l * DD * DD;  const float* bk  = bk_a  + (size_t)l * DD;
    const float* Wv  = Wv_a  + (size_t)l * DD * DD;  const float* bv  = bv_a  + (size_t)l * DD;
    const float* Wqg = Wqg_a + (size_t)l * DD * DD;  const float* bqg = bqg_a + (size_t)l * DD;
    const float* Wkg = Wkg_a + (size_t)l * DD * DD;  const float* bkg = bkg_a + (size_t)l * DD;
    const float* Wvg = Wvg_a + (size_t)l * DD * DD;  const float* bvg = bvg_a + (size_t)l * DD;
    const float* Wo  = Wo_a  + (size_t)l * DD * DD;  const float* bo  = bo_a  + (size_t)l * DD;
    const float* l1g = l1g_a + (size_t)l * DD;       const float* l1b = l1b_a + (size_t)l * DD;
    const float* W1  = W1_a  + (size_t)l * DD * FFD; const float* b1  = b1_a  + (size_t)l * FFD;
    const float* W2  = W2_a  + (size_t)l * FFD * DD; const float* b2  = b2_a  + (size_t)l * DD;
    const float* l2g = l2g_a + (size_t)l * DD;       const float* l2b = l2b_a + (size_t)l * DD;

    // projections (q carries the 1/sqrt(dh) scale)
    gemm_kernel<<<gDxD, blk256, 0, stream>>>(x, Wq,  bq,  qb,  (int)NTOK, DD, DD, 0.125f, 0);
    gemm_kernel<<<gDxD, blk256, 0, stream>>>(x, Wk,  bk,  kb,  (int)NTOK, DD, DD, 1.0f,   0);
    gemm_kernel<<<gDxD, blk256, 0, stream>>>(x, Wv,  bv,  vb,  (int)NTOK, DD, DD, 1.0f,   0);
    gemm_kernel<<<gDxD, blk256, 0, stream>>>(x, Wkg, bkg, kgb, (int)NTOK, DD, DD, 1.0f,   0);
    gemm_kernel<<<gDxD, blk256, 0, stream>>>(x, Wvg, bvg, vgb, (int)NTOK, DD, DD, 1.0f,   0);
    qg_kernel<<<BB, blk256, 0, stream>>>(x, Wqg, bqg, qg);

    // banded attention
    attn_sg_kernel<<<ABLKS, blk256, 0, stream>>>(qb, kb, scores);
    attn_scores_kernel<<<ABLKS, blk256, 0, stream>>>(qb, kb, amask, scores);
    softmax_kernel<<<(int)NROWS, blk256, 0, stream>>>(scores, p0, ph);
    attn_out_kernel<<<ABLKS, blk256, 0, stream>>>(ph, vb, p0, ctx);
    // global token overwrites ctx row 0
    global_attn_kernel<<<BB * HH, blk256, 0, stream>>>(qg, kgb, vgb, amask, ctx);

    // output projection + residual LN
    gemm_kernel<<<gDxD, blk256, 0, stream>>>(ctx, Wo, bo, proj, (int)NTOK, DD, DD, 1.0f, 0);
    add_ln_kernel<<<(int)NTOK, blk256, 0, stream>>>(x, proj, l1g, l1b, x);

    // FFN
    gemm_kernel<<<gDxFF, blk256, 0, stream>>>(x, W1, b1, ffh, (int)NTOK, FFD, DD, 1.0f, 1);
    gemm_kernel<<<gDxD,  blk256, 0, stream>>>(ffh, W2, b2, ffy, (int)NTOK, DD, FFD, 1.0f, 0);
    add_ln_kernel<<<(int)NTOK, blk256, 0, stream>>>(x, ffy, l2g, l2b, x);
  }

  // heads: out layout = stack([police, drug]) -> [2, B]
  head_kernel<<<1, blk256, 0, stream>>>(x, bn_g, bn_b, polW1, polb1, polW2, polb2,
                                        (float*)d_out, 0);
  head_kernel<<<1, blk256, 0, stream>>>(x, bn_g, bn_b, drgW1, drgb1, drgW2, drgb2,
                                        (float*)d_out, 2);
}
